// MPv2_3d_arch_13846974562655
// MI455X (gfx1250) — matrix operand order repaired and checked by reading, not compiled, not run on hardware
//
#include <hip/hip_runtime.h>
#include <hip/hip_bf16.h>

// ---------------------------------------------------------------------------
// CDNA5 (gfx1250) implementation of the ATD transformer block.
// wave32, WMMA f32_16x16x32_f16 for all dense contractions.
// WMMA operand loads (except the per-element V gather) are branch-free aligned vector loads:
//   A fragments: row-major f16, K padded to 32      -> 2x b128 per fragment
//   B fragments: transposed (N-major) f16, padded   -> 2x b128 per fragment
//   qkv for attention: head-padded [3][6][32] f16   -> 2x b128 per fragment
// ---------------------------------------------------------------------------

typedef __attribute__((ext_vector_type(16))) _Float16 v16h;
typedef __attribute__((ext_vector_type(8)))  _Float16 h8;
typedef __attribute__((ext_vector_type(8)))  float    v8f;

union AF { v16h v; _Float16 e[16]; h8 h[2]; };
union CF { v8f  v; float    e[8]; };

#define DEVFN static __device__ __forceinline__

DEVFN float wsum(float s){ for (int o = 16; o; o >>= 1) s += __shfl_xor(s, o, 32); return s; }
DEVFN float wmaxf(float s){ for (int o = 16; o; o >>= 1) s = fmaxf(s, __shfl_xor(s, o, 32)); return s; }
DEVFN float geluf(float x){ return 0.5f * x * (1.0f + erff(x * 0.70710678118654752f)); }

DEVFN v8f wmma_f16(const AF& a, const AF& b, const CF& c) {
    return __builtin_amdgcn_wmma_f32_16x16x32_f16(false, a.v, false, b.v,
                                                  (short)0, c.v, false, false);
}

// ---------------------------------------------------------------------------
// Generic f16 GEMM:  D[.,N](f32) = act(A[.,KP] * Bt^T + bias) + resid
// A: row-major f16, lda = KP (KP % 32 == 0).
// Bt: TRANSPOSED f16 [NPP][KP]; NPP >= gridDim.x*64 so loads are unguarded.
// Optional per-batch Bt (batchRows rows of A share one Bt of strideB halves).
// block = 128 threads = 4 waves; each wave one 16x16 tile of a 16x64 block
// tile; grid = (ceil(N/64), M/16).
// ---------------------------------------------------------------------------
__global__ __launch_bounds__(128) void gemm16(
    const _Float16* __restrict__ A, const _Float16* __restrict__ Bt,
    const float* __restrict__ bias, const float* __restrict__ resid,
    float* __restrict__ D, int N, int ldd, int KP,
    int batchRows, long long strideB, int act)
{
    const int wave = threadIdx.x >> 5, lane = threadIdx.x & 31;
    const int rl = lane & 15, hi = lane >> 4;
    const int row  = blockIdx.y * 16 + rl;
    const int col0 = blockIdx.x * 64 + wave * 16;
    const int batch = (blockIdx.y * 16) / batchRows;

    const _Float16* __restrict__ Ap = A  + (long long)row * KP + hi * 16;
    const _Float16* __restrict__ Bp = Bt + (long long)batch * strideB
                                         + (long long)(col0 + rl) * KP + hi * 16;
    CF acc;
#pragma unroll
    for (int i = 0; i < 8; i++) acc.e[i] = 0.f;

    for (int k0 = 0; k0 < KP; k0 += 32) {
        AF a, b;
        const h8* pa = reinterpret_cast<const h8*>(Ap + k0);
        a.h[0] = pa[0];          // K = hi*16+0..7
        a.h[1] = pa[1];          // K = hi*16+8..15
        const h8* pb = reinterpret_cast<const h8*>(Bp + k0);
        b.h[0] = pb[0];          // K = hi*16+0..7
        b.h[1] = pb[1];          // K = hi*16+8..15
        acc.v = wmma_f16(a, b, acc);
    }

#pragma unroll
    for (int vv = 0; vv < 8; vv++) {
        int m = blockIdx.y * 16 + vv + hi * 8;
        int n = col0 + rl;
        if (n < N) {
            float val = acc.e[vv] + (bias ? bias[n] : 0.f);
            if (act) val = geluf(val);
            if (resid) val += resid[(long long)m * ldd + n];
            D[(long long)m * ldd + n] = val;
        }
    }
}

// ---------------------------------------------------------------------------
// f32 -> padded row-major f16 (for A-side matrices)
// ---------------------------------------------------------------------------
__global__ void pack_f16(const float* __restrict__ src, _Float16* __restrict__ dst,
                         int K, int N, int KP, int NP)
{
    long long i = (long long)blockIdx.x * 256 + threadIdx.x;
    if (i >= (long long)KP * NP) return;
    int n = (int)(i % NP), k = (int)(i / NP);
    dst[i] = (k < K && n < N) ? (_Float16)src[(long long)k * N + n] : (_Float16)0;
}

// f32 [nb][K][N] -> transposed padded f16 [nb][NPP][KP]  (for B-side)
__global__ void pack_tr_f16(const float* __restrict__ src, _Float16* __restrict__ dst,
                            int K, int N, int KP, int NPP, int nb)
{
    long long i = (long long)blockIdx.x * 256 + threadIdx.x;
    if (i >= (long long)nb * NPP * KP) return;
    int k = (int)(i % KP);
    long long r = i / KP;
    int n = (int)(r % NPP);
    int b = (int)(r / NPP);
    dst[i] = (k < K && n < N)
        ? (_Float16)src[((long long)b * K + k) * N + n] : (_Float16)0;
}

// f32 [180][N] -> transposed f16 [NPP][192] with head-padded K (6 heads x 30->32)
__global__ void pack_tr_hp_f16(const float* __restrict__ src, _Float16* __restrict__ dst,
                               int N, int NPP)
{
    long long i = (long long)blockIdx.x * 256 + threadIdx.x;
    if (i >= (long long)NPP * 192) return;
    int k = (int)(i % 192);
    int n = (int)(i / 192);
    int h = k >> 5, d = k & 31;
    dst[i] = (d < 30 && n < N)
        ? (_Float16)src[(long long)(h * 30 + d) * N + n] : (_Float16)0;
}

// qkv f32 [row][540] -> head-padded f16 [row][3][6][32]; q section pre-scaled
__global__ void repack_qkv(const float* __restrict__ qkv, _Float16* __restrict__ q16)
{
    long long i = (long long)blockIdx.x * 256 + threadIdx.x;
    if (i >= (long long)32768 * 576) return;
    long long row = i / 576;
    int c = (int)(i % 576);
    int s = c / 192, r = c % 192, h = r >> 5, d = r & 31;
    float v = (d < 30) ? qkv[row * 540 + s * 180 + h * 30 + d] : 0.f;
    if (s == 0) v *= 0.18257418583505537f;          // hd^-0.5, hd = 30
    q16[i] = (_Float16)v;
}

__global__ void zero_cols_f16(_Float16* __restrict__ p, long long rows, int ld, int c0, int nc)
{
    long long i = (long long)blockIdx.x * 256 + threadIdx.x;
    if (i >= rows * nc) return;
    long long r = i / nc; int c = c0 + (int)(i % nc);
    p[r * ld + c] = (_Float16)0;
}

// ---------------------------------------------------------------------------
// LayerNorm over 180 channels; one wave per row. Optional f32 copy plus a
// zero-padded f16 copy (ld16 = 192) for the WMMA GEMMs.
// ---------------------------------------------------------------------------
__global__ __launch_bounds__(32) void ln_rows(
    const float* __restrict__ x, const float* __restrict__ g, const float* __restrict__ b,
    float* __restrict__ y32, _Float16* __restrict__ y16, int ld16)
{
    long long row = blockIdx.x;
    int lane = threadIdx.x;
    float v[6]; float s = 0.f;
#pragma unroll
    for (int j = 0; j < 6; j++) {
        int c = lane + 32 * j;
        v[j] = (c < 180) ? x[row * 180 + c] : 0.f;
        s += v[j];
    }
    s = wsum(s);
    float mean = s * (1.f / 180.f);
    float s2 = 0.f;
#pragma unroll
    for (int j = 0; j < 6; j++) {
        int c = lane + 32 * j;
        if (c < 180) { float d = v[j] - mean; s2 += d * d; }
    }
    s2 = wsum(s2);
    float inv = rsqrtf(s2 * (1.f / 180.f) + 1e-5f);
#pragma unroll
    for (int j = 0; j < 6; j++) {
        int c = lane + 32 * j;
        if (c < 180) {
            float o = (v[j] - mean) * inv * g[c] + b[c];
            if (y32) y32[row * 180 + c] = o;
            y16[row * ld16 + c] = (_Float16)o;
        } else if (c < ld16) {
            y16[row * ld16 + c] = (_Float16)0;
        }
    }
}

// ---------------------------------------------------------------------------
// y = l2norm(X[rows,180] @ W[180,10] + b)  (ATD q / k projections)
// ---------------------------------------------------------------------------
__global__ __launch_bounds__(256) void l2proj10(
    const float* __restrict__ X, const float* __restrict__ W,
    const float* __restrict__ B, float* __restrict__ out, int rows)
{
    __shared__ float wl[1800];
    __shared__ float bl[10];
    for (int i = threadIdx.x; i < 1800; i += 256) wl[i] = W[i];
    if (threadIdx.x < 10) bl[threadIdx.x] = B[threadIdx.x];
    __syncthreads();
    int row = blockIdx.x * 256 + threadIdx.x;
    if (row >= rows) return;
    float acc[10];
#pragma unroll
    for (int r = 0; r < 10; r++) acc[r] = bl[r];
    for (int d = 0; d < 180; d++) {
        float xv = X[(long long)row * 180 + d];
#pragma unroll
        for (int r = 0; r < 10; r++) acc[r] += xv * wl[d * 10 + r];
    }
    float n2 = 0.f;
#pragma unroll
    for (int r = 0; r < 10; r++) n2 += acc[r] * acc[r];
    float inv = 1.f / fmaxf(sqrtf(n2), 1e-12f);
#pragma unroll
    for (int r = 0; r < 10; r++) out[(long long)row * 10 + r] = acc[r] * inv;
}

// ---------------------------------------------------------------------------
// ATD cross-attn softmax: probs (f16) + argmax token id.  One thread per row.
// ---------------------------------------------------------------------------
__global__ __launch_bounds__(256) void atd_sim(
    const float* __restrict__ qn, const float* __restrict__ kkn,
    const float* __restrict__ sc, _Float16* __restrict__ P,
    int* __restrict__ tk)
{
    const int batch = blockIdx.x / 16;           // 16 blocks of 256 rows per batch
    __shared__ float kl[640];
    for (int i = threadIdx.x; i < 640; i += 256) kl[i] = kkn[batch * 640 + i];
    __syncthreads();
    long long row = (long long)blockIdx.x * 256 + threadIdx.x;
    float q[10];
#pragma unroll
    for (int r = 0; r < 10; r++) q[r] = qn[row * 10 + r];
    float scale = 1.f + fminf(fmaxf(sc[0], 0.f), 3.f) * logf(64.f);
    float s[64]; float mx = -1e30f; int am = 0;
#pragma unroll
    for (int m = 0; m < 64; m++) {
        float d = 0.f;
#pragma unroll
        for (int r = 0; r < 10; r++) d += q[r] * kl[m * 10 + r];
        d *= scale; s[m] = d;
        if (d > mx) { mx = d; am = m; }
    }
    float se = 0.f;
#pragma unroll
    for (int m = 0; m < 64; m++) { s[m] = __expf(s[m] - mx); se += s[m]; }
    float inv = 1.f / se;
#pragma unroll
    for (int m = 0; m < 64; m++) P[row * 64 + m] = (_Float16)(s[m] * inv);
    tk[row] = am;
}

// ---------------------------------------------------------------------------
// Stable counting sort of tk_id (64 bins) per batch -> sort_idx (== argsort).
// ---------------------------------------------------------------------------
__global__ __launch_bounds__(64) void sort_tk(const int* __restrict__ tk, int* __restrict__ sidx)
{
    const int b = blockIdx.x, t = threadIdx.x;
    __shared__ int tl[4096];
    __shared__ int cnt[64];
    __shared__ int ofs[64];
    for (int i = t; i < 4096; i += 64) tl[i] = tk[b * 4096 + i];
    __syncthreads();
    int c = 0;
    for (int i = 0; i < 4096; i++) c += (tl[i] == t);
    cnt[t] = c;
    __syncthreads();
    if (t == 0) { int s = 0; for (int j = 0; j < 64; j++) { ofs[j] = s; s += cnt[j]; } }
    __syncthreads();
    int p = ofs[t];
    for (int i = 0; i < 4096; i++)
        if (tl[i] == t) sidx[b * 4096 + (p++)] = i;
}

// ---------------------------------------------------------------------------
// Window attention (16x16 windows, 6 heads, hd 30->32 padded, N=256) + RPB.
// One wave per (window, head, 16-row tile). WMMA scores + WMMA P*V.
// Reads head-padded qkv16 [row][3][6][32]; writes head-padded f16 [row][192].
// ---------------------------------------------------------------------------
__global__ __launch_bounds__(32) void win_attn(
    const _Float16* __restrict__ q16, const int* __restrict__ rpi,
    const float* __restrict__ rpb, _Float16* __restrict__ out)
{
    int id = blockIdx.x;
    const int rt = id & 15; id >>= 4;
    const int head = id % 6; id /= 6;
    const int win = id;                    // 0..127
    const int b = win >> 4, wloc = win & 15;
    const int wy = wloc >> 2, wx = wloc & 3;
    const int lane = threadIdx.x, rl = lane & 15, hi = lane >> 4;

    __shared__ float    S[16][264];
    __shared__ _Float16 Pp[16][256];

    auto nIdx = [&](int i) { return ((wy * 16 + (i >> 4)) << 6) + wx * 16 + (i & 15); };
    const long long base = (long long)b * 4096;
    const int hc = head * 32;

    AF Aq;
    {
        const h8* pq = reinterpret_cast<const h8*>(
            q16 + (base + nIdx(rt * 16 + rl)) * 576 + hc + hi * 16);
        Aq.h[0] = pq[0]; Aq.h[1] = pq[1];
    }
    for (int ct = 0; ct < 16; ct++) {
        AF Bk;
        const h8* pk = reinterpret_cast<const h8*>(
            q16 + (base + nIdx(ct * 16 + rl)) * 576 + 192 + hc + hi * 16);
        Bk.h[0] = pk[0]; Bk.h[1] = pk[1];
        CF acc;
#pragma unroll
        for (int i = 0; i < 8; i++) acc.e[i] = 0.f;
        acc.v = wmma_f16(Aq, Bk, acc);
#pragma unroll
        for (int vv = 0; vv < 8; vv++) {
            int m = vv + hi * 8;
            int iL = rt * 16 + m, jL = ct * 16 + rl;
            float bias = rpb[rpi[iL * 256 + jL] * 6 + head];
            S[m][ct * 16 + rl] = acc.e[vv] + bias;
        }
    }
    __syncthreads();
    for (int r = 0; r < 16; r++) {
        float mx = -1e30f;
        for (int c = lane; c < 256; c += 32) mx = fmaxf(mx, S[r][c]);
        mx = wmaxf(mx);
        float se = 0.f;
        for (int c = lane; c < 256; c += 32) { float e = __expf(S[r][c] - mx); S[r][c] = e; se += e; }
        se = wsum(se);
        float inv = 1.f / se;
        for (int c = lane; c < 256; c += 32) Pp[r][c] = (_Float16)(S[r][c] * inv);
    }
    __syncthreads();
    // O = P * V  (two 16-wide column tiles; cols 30,31 are exactly zero)
    for (int t = 0; t < 2; t++) {
        CF acc;
#pragma unroll
        for (int i = 0; i < 8; i++) acc.e[i] = 0.f;
        const int d = t * 16 + rl;
        for (int kt = 0; kt < 8; kt++) {
            AF Ap;
            const h8* pp = reinterpret_cast<const h8*>(&Pp[rl][kt * 32 + hi * 16]);
            Ap.h[0] = pp[0]; Ap.h[1] = pp[1];
            AF Bv;
#pragma unroll
            for (int i = 0; i < 16; i++) {
                int j = kt * 32 + hi * 16 + i;
                Bv.e[i] = q16[(base + nIdx(j)) * 576 + 384 + hc + d];
            }
            acc.v = wmma_f16(Ap, Bv, acc);
        }
#pragma unroll
        for (int vv = 0; vv < 8; vv++) {
            int m = vv + hi * 8;
            out[(base + nIdx(rt * 16 + m)) * 192 + hc + t * 16 + rl] = (_Float16)acc.e[vv];
        }
    }
}

// ---------------------------------------------------------------------------
// AC_MSA: category-sorted grouped attention (groups of 128, 6 heads).
// Rows gathered through sort_idx, output scattered back through sort_idx
// (equivalent to the inv_idx un-permute in the reference).
// ---------------------------------------------------------------------------
__global__ __launch_bounds__(32) void aca_attn(
    const _Float16* __restrict__ q16, const int* __restrict__ sidx,
    _Float16* __restrict__ out)
{
    int id = blockIdx.x;
    const int rt = id & 7; id >>= 3;
    const int head = id % 6; id /= 6;
    const int g = id & 31; const int b = id >> 5;
    const int lane = threadIdx.x, rl = lane & 15, hi = lane >> 4;

    __shared__ float    S[16][136];
    __shared__ _Float16 Pp[16][128];
    __shared__ int      nmap[128];

    const long long base = (long long)b * 4096;
    const int hc = head * 32;
    for (int i = lane; i < 128; i += 32) nmap[i] = sidx[base + g * 128 + i];
    __syncthreads();

    AF Aq;
    {
        const h8* pq = reinterpret_cast<const h8*>(
            q16 + (base + nmap[rt * 16 + rl]) * 576 + hc + hi * 16);
        Aq.h[0] = pq[0]; Aq.h[1] = pq[1];
    }
    for (int ct = 0; ct < 8; ct++) {
        AF Bk;
        const h8* pk = reinterpret_cast<const h8*>(
            q16 + (base + nmap[ct * 16 + rl]) * 576 + 192 + hc + hi * 16);
        Bk.h[0] = pk[0]; Bk.h[1] = pk[1];
        CF acc;
#pragma unroll
        for (int i = 0; i < 8; i++) acc.e[i] = 0.f;
        acc.v = wmma_f16(Aq, Bk, acc);
#pragma unroll
        for (int vv = 0; vv < 8; vv++) S[vv + hi * 8][ct * 16 + rl] = acc.e[vv];
    }
    __syncthreads();
    for (int r = 0; r < 16; r++) {
        float mx = -1e30f;
        for (int c = lane; c < 128; c += 32) mx = fmaxf(mx, S[r][c]);
        mx = wmaxf(mx);
        float se = 0.f;
        for (int c = lane; c < 128; c += 32) { float e = __expf(S[r][c] - mx); S[r][c] = e; se += e; }
        se = wsum(se);
        float inv = 1.f / se;
        for (int c = lane; c < 128; c += 32) Pp[r][c] = (_Float16)(S[r][c] * inv);
    }
    __syncthreads();
    for (int t = 0; t < 2; t++) {
        CF acc;
#pragma unroll
        for (int i = 0; i < 8; i++) acc.e[i] = 0.f;
        const int d = t * 16 + rl;
        for (int kt = 0; kt < 4; kt++) {
            AF Ap;
            const h8* pp = reinterpret_cast<const h8*>(&Pp[rl][kt * 32 + hi * 16]);
            Ap.h[0] = pp[0]; Ap.h[1] = pp[1];
            AF Bv;
#pragma unroll
            for (int i = 0; i < 16; i++) {
                int j = kt * 32 + hi * 16 + i;
                Bv.e[i] = q16[(base + nmap[j]) * 576 + 384 + hc + d];
            }
            acc.v = wmma_f16(Ap, Bv, acc);
        }
#pragma unroll
        for (int vv = 0; vv < 8; vv++) {
            int m = vv + hi * 8;
            out[(base + nmap[rt * 16 + m]) * 192 + hc + t * 16 + rl] = (_Float16)acc.e[vv];
        }
    }
}

// ---------------------------------------------------------------------------
// hc[:, 360:424] = td_proj[b, tk_id[row], :]
// ---------------------------------------------------------------------------
__global__ __launch_bounds__(256) void gather_td(
    const float* __restrict__ td_proj, const int* __restrict__ tk,
    float* __restrict__ hc)
{
    long long idx = (long long)blockIdx.x * 256 + threadIdx.x;
    long long row = idx >> 6; int c = (int)(idx & 63);
    int b = (int)(row >> 12);
    hc[row * 424 + 360 + c] = td_proj[((long long)b * 64 + tk[row]) * 64 + c];
}

// ---------------------------------------------------------------------------
// 5x5 depthwise conv (SAME) over 64x64 planes + GELU + residual; emits the
// f16 (ld 448) input for the fc2 WMMA GEMM. One block per (batch, channel).
// ---------------------------------------------------------------------------
__global__ __launch_bounds__(256) void dwconv(
    const float* __restrict__ hc, const float* __restrict__ w,
    const float* __restrict__ bb, _Float16* __restrict__ out)
{
    const int b = blockIdx.x / 424;
    const int c = blockIdx.x % 424;
    __shared__ float img[68 * 68];
    __shared__ float wv[25];
    for (int i = threadIdx.x; i < 68 * 68; i += 256) img[i] = 0.f;
    if (threadIdx.x < 25) wv[threadIdx.x] = w[c * 25 + threadIdx.x];
    __syncthreads();
    for (int p = threadIdx.x; p < 4096; p += 256)
        img[((p >> 6) + 2) * 68 + (p & 63) + 2] = hc[((long long)b * 4096 + p) * 424 + c];
    __syncthreads();
    const float bias = bb[c];
    for (int p = threadIdx.x; p < 4096; p += 256) {
        int y = p >> 6, x = p & 63;
        float a = 0.f;
#pragma unroll
        for (int ky = 0; ky < 5; ky++)
#pragma unroll
            for (int kx = 0; kx < 5; kx++)
                a += img[(y + ky) * 68 + x + kx] * wv[ky * 5 + kx];
        float center = img[(y + 2) * 68 + x + 2];
        float o = center + geluf(a + bias);
        out[((long long)b * 4096 + p) * 448 + c] = (_Float16)o;
    }
}

// ---------------------------------------------------------------------------
// Host-side orchestration
// ---------------------------------------------------------------------------
extern "C" void kernel_launch(void* const* d_in, const int* in_sizes, int n_in,
                              void* d_out, int out_size, void* d_ws, size_t ws_size,
                              hipStream_t stream)
{
    (void)in_sizes; (void)n_in; (void)out_size; (void)ws_size;
    const float* d_x       = (const float*)d_in[0];
    const float* d_td      = (const float*)d_in[1];
    const float* d_n1g     = (const float*)d_in[2];
    const float* d_n1b     = (const float*)d_in[3];
    const float* d_wqkv_w  = (const float*)d_in[4];
    const float* d_wqkv_b  = (const float*)d_in[5];
    const float* d_atdq_w  = (const float*)d_in[6];
    const float* d_atdq_b  = (const float*)d_in[7];
    const float* d_atdk_w  = (const float*)d_in[8];
    const float* d_atdk_b  = (const float*)d_in[9];
    const float* d_atdv_w  = (const float*)d_in[10];
    const float* d_atdv_b  = (const float*)d_in[11];
    const float* d_scale   = (const float*)d_in[12];
    const float* d_aca_w   = (const float*)d_in[13];
    const float* d_aca_b   = (const float*)d_in[14];
    const float* d_rpb     = (const float*)d_in[15];
    const float* d_win_w   = (const float*)d_in[16];
    const float* d_win_b   = (const float*)d_in[17];
    const float* d_fctd_w  = (const float*)d_in[18];
    const float* d_fctd_b  = (const float*)d_in[19];
    const float* d_fc1_w   = (const float*)d_in[20];
    const float* d_fc1_b   = (const float*)d_in[21];
    const float* d_dw_w    = (const float*)d_in[22];
    const float* d_dw_b    = (const float*)d_in[23];
    const float* d_fc2_w   = (const float*)d_in[24];
    const float* d_fc2_b   = (const float*)d_in[25];
    const float* d_n2g     = (const float*)d_in[26];
    const float* d_n2b     = (const float*)d_in[27];
    const int*   d_rpi     = (const int*)d_in[28];
    float*       out       = (float*)d_out;

    // workspace carve-up (256B aligned)
    char* wp = (char*)d_ws;
    auto alloc = [&](size_t bytes) { char* p = wp; wp += (bytes + 255) & ~(size_t)255; return p; };
    const long long BT = 32768;                  // B * n

    _Float16* Wqkv16  = (_Float16*)alloc(576 * 192 * 2);   // transposed
    _Float16* Waca16  = (_Float16*)alloc(192 * 192 * 2);   // transposed, head-padded K
    _Float16* Wwin16  = (_Float16*)alloc(192 * 192 * 2);   // transposed, head-padded K
    _Float16* Watdv16 = (_Float16*)alloc(192 * 192 * 2);   // transposed
    _Float16* Wfc1_16 = (_Float16*)alloc(384 * 192 * 2);   // transposed
    _Float16* Wfc2_16 = (_Float16*)alloc(192 * 448 * 2);   // transposed
    _Float16* Wfctd16 = (_Float16*)alloc(64 * 192 * 2);    // transposed
    _Float16* vv16t   = (_Float16*)alloc(8 * 192 * 64 * 2);// transposed, batched
    _Float16* td16    = (_Float16*)alloc(512 * 192 * 2);   // row-major (A side)
    _Float16* Xn16    = (_Float16*)alloc(BT * 192 * 2);
    _Float16* X2_16   = (_Float16*)alloc(BT * 192 * 2);
    _Float16* Xw16    = (_Float16*)alloc(BT * 192 * 2);    // head-padded
    _Float16* Yaca16  = (_Float16*)alloc(BT * 192 * 2);    // head-padded
    _Float16* P16     = (_Float16*)alloc(BT * 64 * 2);
    _Float16* QKV16   = (_Float16*)alloc(BT * 576 * 2);    // head-padded qkv
    _Float16* HC16    = (_Float16*)alloc(BT * 448 * 2);
    float* xn    = (float*)alloc(BT * 180 * 4);
    float* qkv   = (float*)alloc(BT * 540 * 4);
    float* X1    = (float*)alloc(BT * 180 * 4);
    float* hc32  = (float*)alloc(BT * 424 * 4);
    float* vv    = (float*)alloc(512 * 180 * 4);
    float* qn    = (float*)alloc(BT * 10 * 4);
    float* kkn   = (float*)alloc(512 * 10 * 4);
    float* tdp   = (float*)alloc(512 * 64 * 4);
    int*   tk    = (int*)alloc(BT * 4);
    int*   sidx  = (int*)alloc(BT * 4);

    const int BIG = 0x3FFFFFFF;
    auto pgrid = [](long long n) { return (unsigned)((n + 255) / 256); };

    // --- pack weights (transposed B-side), td (A-side) ---
    pack_tr_f16<<<pgrid(576 * 192), 256, 0, stream>>>(d_wqkv_w, Wqkv16, 180, 540, 192, 576, 1);
    pack_tr_hp_f16<<<pgrid(192 * 192), 256, 0, stream>>>(d_aca_w, Waca16, 180, 192);
    pack_tr_hp_f16<<<pgrid(192 * 192), 256, 0, stream>>>(d_win_w, Wwin16, 180, 192);
    pack_tr_f16<<<pgrid(192 * 192), 256, 0, stream>>>(d_atdv_w, Watdv16, 180, 180, 192, 192, 1);
    pack_tr_f16<<<pgrid(384 * 192), 256, 0, stream>>>(d_fc1_w, Wfc1_16, 180, 360, 192, 384, 1);
    pack_tr_f16<<<pgrid(192 * 448), 256, 0, stream>>>(d_fc2_w, Wfc2_16, 424, 180, 448, 192, 1);
    pack_tr_f16<<<pgrid(64 * 192),  256, 0, stream>>>(d_fctd_w, Wfctd16, 180, 64, 192, 64, 1);
    pack_f16<<<pgrid(512 * 192), 256, 0, stream>>>(d_td, td16, 512, 180, 512, 192);

    // --- LN1 ---
    ln_rows<<<32768, 32, 0, stream>>>(d_x, d_n1g, d_n1b, xn, Xn16, 192);

    // --- qkv projection, then head-padded f16 repack for attention ---
    gemm16<<<dim3(9, 2048), 128, 0, stream>>>(Xn16, Wqkv16, d_wqkv_b, nullptr, qkv,
                                              540, 540, 192, BIG, 0, 0);
    repack_qkv<<<pgrid(BT * 576), 256, 0, stream>>>(qkv, QKV16);

    // --- ATD cross attention ---
    l2proj10<<<128, 256, 0, stream>>>(xn, d_atdq_w, d_atdq_b, qn, 32768);
    l2proj10<<<2, 256, 0, stream>>>(d_td, d_atdk_w, d_atdk_b, kkn, 512);
    gemm16<<<dim3(3, 32), 128, 0, stream>>>(td16, Watdv16, d_atdv_b, nullptr, vv,
                                            180, 180, 192, BIG, 0, 0);
    pack_tr_f16<<<pgrid(8 * 192 * 64), 256, 0, stream>>>(vv, vv16t, 64, 180, 64, 192, 8);
    atd_sim<<<128, 256, 0, stream>>>(qn, kkn, d_scale, P16, tk);
    // X1 = x + attn_atd @ vv   (per-batch 192x64 transposed slice of vv16t)
    gemm16<<<dim3(3, 2048), 128, 0, stream>>>(P16, vv16t, nullptr, d_x, X1,
                                              180, 180, 64, 4096, 192 * 64, 0);

    // --- window attention;  X1 += win_proj(ow) ---
    win_attn<<<12288, 32, 0, stream>>>(QKV16, d_rpi, d_rpb, Xw16);
    gemm16<<<dim3(3, 2048), 128, 0, stream>>>(Xw16, Wwin16, d_win_b, X1, X1,
                                              180, 180, 192, BIG, 0, 0);

    // --- AC_MSA;  X1 += aca_proj(y) ---
    sort_tk<<<8, 64, 0, stream>>>(tk, sidx);
    aca_attn<<<12288, 32, 0, stream>>>(QKV16, sidx, Yaca16);
    gemm16<<<dim3(3, 2048), 128, 0, stream>>>(Yaca16, Waca16, d_aca_b, X1, X1,
                                              180, 180, 192, BIG, 0, 0);

    // --- ConvFFN_td ---
    gemm16<<<dim3(1, 32), 128, 0, stream>>>(td16, Wfctd16, d_fctd_b, nullptr, tdp,
                                            64, 64, 192, BIG, 0, 0);
    ln_rows<<<32768, 32, 0, stream>>>(X1, d_n2g, d_n2b, nullptr, X2_16, 192);
    gemm16<<<dim3(6, 2048), 128, 0, stream>>>(X2_16, Wfc1_16, d_fc1_b, nullptr, hc32,
                                              360, 424, 192, BIG, 0, 1 /*gelu*/);
    gather_td<<<8192, 256, 0, stream>>>(tdp, tk, hc32);
    dwconv<<<8 * 424, 256, 0, stream>>>(hc32, d_dw_w, d_dw_b, HC16);
    zero_cols_f16<<<pgrid(BT * 24), 256, 0, stream>>>(HC16, BT, 448, 424, 24);
    // out = X1 + hc_new @ fc2_w + fc2_b
    gemm16<<<dim3(3, 2048), 128, 0, stream>>>(HC16, Wfc2_16, d_fc2_b, X1, out,
                                              180, 180, 448, BIG, 0, 0);
}
